// sparse_moe_56650618634405
// MI455X (gfx1250) — matrix operand order repaired and checked by reading, not compiled, not run on hardware
//
#include <hip/hip_runtime.h>
#include <hip/hip_bf16.h>

// ---------------- problem constants (from reference setup) ----------------
#define NTOK 2048      // B*T = 2*1024
#define DDIM 768
#define NEXP 8
#define HDIM 3072
#define MTILE 32       // tokens per expert-kernel block (2 WMMA row-tiles)

typedef __attribute__((ext_vector_type(16))) __bf16 v16bf;
typedef __attribute__((ext_vector_type(8)))  float  v8f;

// ---------------- ws layout ----------------
#define XB_BYTES   ((size_t)NTOK * DDIM * 2)
#define W1T_BYTES  ((size_t)NEXP * HDIM * DDIM * 2)
#define W2T_BYTES  ((size_t)NEXP * DDIM * HDIM * 2)
#define OFF_XB     ((size_t)0)
#define OFF_W1T    (OFF_XB + XB_BYTES)
#define OFF_W2T    (OFF_W1T + W1T_BYTES)
#define OFF_CNT    (OFF_W2T + W2T_BYTES)
#define OFF_LIST   (OFF_CNT + 256)
#define OFF_GVAL   (OFF_LIST + (size_t)NEXP * NTOK * 4)

// ---------------- kernel 0: zero output + counters ----------------
__global__ __launch_bounds__(256)
void moe_zero_kernel(float* __restrict__ out, int* __restrict__ cnt) {
  int idx = blockIdx.x * blockDim.x + threadIdx.x;
  const int n = NTOK * DDIM;
  for (int i = idx; i < n; i += gridDim.x * blockDim.x) out[i] = 0.0f;
  if (idx < NEXP) cnt[idx] = 0;
}

// ---------------- kernel 1: per-expert transpose + fp32 -> bf16 ----------
// src: [E][R][C] f32 row-major;  dst: [E][C][R] bf16 row-major
__global__ __launch_bounds__(256)
void moe_transpose_bf16_kernel(const float* __restrict__ src,
                               __hip_bfloat16* __restrict__ dst,
                               int R, int C) {
  __shared__ float tile[32][33];
  const int e = blockIdx.z;
  src += (size_t)e * R * C;
  dst += (size_t)e * R * C;
  const int c0 = blockIdx.x * 32;
  const int r0 = blockIdx.y * 32;
#pragma unroll
  for (int i = threadIdx.y; i < 32; i += 8)
    tile[i][threadIdx.x] = src[(size_t)(r0 + i) * C + (c0 + threadIdx.x)];
  __syncthreads();
#pragma unroll
  for (int i = threadIdx.y; i < 32; i += 8)
    dst[(size_t)(c0 + i) * R + (r0 + threadIdx.x)] =
        __float2bfloat16(tile[threadIdx.x][i]);
}

// ---------------- kernel 2: noisy top-2 router + compaction -------------
__global__ __launch_bounds__(256)
void moe_router_kernel(const float* __restrict__ x,
                       const float* __restrict__ noise,
                       const float* __restrict__ w_gate,
                       const float* __restrict__ b_gate,
                       const float* __restrict__ w_noise,
                       const float* __restrict__ b_noise,
                       __hip_bfloat16* __restrict__ xb,
                       int* __restrict__ cnt,
                       int* __restrict__ list,
                       float* __restrict__ gval) {
  const int wave = threadIdx.x >> 5;
  const int lane = threadIdx.x & 31;
  const int token = blockIdx.x * 8 + wave;
  if (token >= NTOK) return;

  float ag[8] = {0,0,0,0,0,0,0,0};
  float an[8] = {0,0,0,0,0,0,0,0};
  const float* xrow = x + (size_t)token * DDIM;
  for (int d = lane; d < DDIM; d += 32) {
    float xv = xrow[d];
    xb[(size_t)token * DDIM + d] = __float2bfloat16(xv);
    const float4* wg = (const float4*)(w_gate + (size_t)d * NEXP);
    const float4* wn = (const float4*)(w_noise + (size_t)d * NEXP);
    float4 g0 = wg[0], g1 = wg[1], n0 = wn[0], n1 = wn[1];
    ag[0] += xv * g0.x; ag[1] += xv * g0.y; ag[2] += xv * g0.z; ag[3] += xv * g0.w;
    ag[4] += xv * g1.x; ag[5] += xv * g1.y; ag[6] += xv * g1.z; ag[7] += xv * g1.w;
    an[0] += xv * n0.x; an[1] += xv * n0.y; an[2] += xv * n0.z; an[3] += xv * n0.w;
    an[4] += xv * n1.x; an[5] += xv * n1.y; an[6] += xv * n1.z; an[7] += xv * n1.w;
  }
#pragma unroll
  for (int off = 16; off > 0; off >>= 1) {
#pragma unroll
    for (int e = 0; e < 8; ++e) {
      ag[e] += __shfl_xor(ag[e], off, 32);
      an[e] += __shfl_xor(an[e], off, 32);
    }
  }
  if (lane == 0) {
    float nv[8];
#pragma unroll
    for (int e = 0; e < 8; ++e) {
      float lg = ag[e] + b_gate[e];
      float nl = an[e] + b_noise[e];
      float sp = (nl > 20.f) ? nl : log1pf(expf(nl));  // softplus
      nv[e] = lg + noise[(size_t)token * NEXP + e] * sp;
    }
    float v1 = -1e38f, v2 = -1e38f; int i1 = 0, i2 = 1;
#pragma unroll
    for (int e = 0; e < 8; ++e) {
      float v = nv[e];
      if (v > v1) { v2 = v1; i2 = i1; v1 = v; i1 = e; }
      else if (v > v2) { v2 = v; i2 = e; }
    }
    float e2 = expf(v2 - v1);
    float inv = 1.0f / (1.0f + e2);
    int p1 = atomicAdd(&cnt[i1], 1);
    list[i1 * NTOK + p1] = token; gval[i1 * NTOK + p1] = inv;
    int p2 = atomicAdd(&cnt[i2], 1);
    list[i2 * NTOK + p2] = token; gval[i2 * NTOK + p2] = e2 * inv;
  }
}

// ---------------- kernel 3: fused expert MLP (WMMA bf16, M=32) ----------
// grid = (NTOK/32, NEXP); block = 256 (8 waves).
// LDS: h tile 32x3072 bf16 (192KB) + A tile 32x768 bf16 (48KB) + gates/toks.
// Each wave computes 2 row-subtiles per N-tile -> every B fragment reused 2x.
__global__ __launch_bounds__(256)
void moe_expert_kernel(const __hip_bfloat16* __restrict__ xb,
                       const __hip_bfloat16* __restrict__ w1t,  // [E][H][D]
                       const __hip_bfloat16* __restrict__ w2t,  // [E][D][H]
                       const float* __restrict__ b1,            // [E][H]
                       const float* __restrict__ b2,            // [E][D]
                       const int* __restrict__ cnt,
                       const int* __restrict__ list,
                       const float* __restrict__ gval,
                       float* __restrict__ out) {
  extern __shared__ char smem[];
  __hip_bfloat16* hT = (__hip_bfloat16*)smem;                 // [32][HDIM]
  __hip_bfloat16* aT = hT + MTILE * HDIM;                     // [32][DDIM]
  float* gts = (float*)(aT + MTILE * DDIM);                   // [32]
  int*   toks = (int*)(gts + MTILE);                          // [32]

  const int e = blockIdx.y;
  const int count = cnt[e];
  const int m0 = blockIdx.x * MTILE;
  if (m0 >= count) return;

  const int tid = threadIdx.x;
  if (tid < MTILE) {
    int idx = m0 + tid;
    bool ok = idx < count;
    toks[tid] = ok ? list[e * NTOK + idx] : -1;
    gts[tid]  = ok ? gval[e * NTOK + idx] : 0.0f;
  }
  __syncthreads();

  // stage A tile (32 x 768 bf16) into LDS, zero-padding invalid rows
  {
    const int chunksPerRow = DDIM / 8;  // 16B chunks
    for (int c = tid; c < MTILE * chunksPerRow; c += 256) {
      int row = c / chunksPerRow, col8 = c % chunksPerRow;
      int tok = toks[row];
      uint4 v = make_uint4(0u, 0u, 0u, 0u);
      if (tok >= 0)
        v = *(const uint4*)(xb + (size_t)tok * DDIM + col8 * 8);
      *(uint4*)(aT + (size_t)row * DDIM + col8 * 8) = v;
    }
  }
  __syncthreads();

  const int wave = tid >> 5;
  const int lane = tid & 31;
  const int nl   = lane & 15;       // column within 16-wide tile (B/C); row for A
  const int hi   = lane >> 4;       // lane half
  const int koffA = hi * 16;        // A-fragment K offset for this half
  const int khalfB = hi * 16;       // B-fragment K offset for this half

  union AF { uint4 q[2]; v16bf v; };

  // ---- layer 1: h = relu(x @ w1 + b1), streamed into LDS ----
  const __hip_bfloat16* w1e = w1t + (size_t)e * HDIM * DDIM;
  for (int t = wave; t < HDIM / 16; t += 8) {
    const int hcol = t * 16;
    float bias = b1[(size_t)e * HDIM + hcol + nl];
    v8f acc0, acc1;
#pragma unroll
    for (int r = 0; r < 8; ++r) { acc0[r] = bias; acc1[r] = bias; }
    const __hip_bfloat16* bcol = w1e + (size_t)(hcol + nl) * DDIM;
    const __hip_bfloat16* arow0 = aT + (size_t)nl * DDIM;
    const __hip_bfloat16* arow1 = aT + (size_t)(16 + nl) * DDIM;
    for (int kb = 0; kb < DDIM; kb += 32) {
      v16bf Bv = *(const v16bf*)(bcol + kb + khalfB);
      __builtin_prefetch(bcol + kb + 64 + khalfB, 0, 1);
      AF A0, A1;
      A0.q[0] = *(const uint4*)(arow0 + kb + koffA);
      A0.q[1] = *(const uint4*)(arow0 + kb + 8 + koffA);
      A1.q[0] = *(const uint4*)(arow1 + kb + koffA);
      A1.q[1] = *(const uint4*)(arow1 + kb + 8 + koffA);
      acc0 = __builtin_amdgcn_wmma_f32_16x16x32_bf16(
          false, A0.v, false, Bv, (short)0, acc0, false, false);
      acc1 = __builtin_amdgcn_wmma_f32_16x16x32_bf16(
          false, A1.v, false, Bv, (short)0, acc1, false, false);
    }
#pragma unroll
    for (int r = 0; r < 8; ++r) {
      float v0 = acc0[r] > 0.0f ? acc0[r] : 0.0f;  // relu
      float v1 = acc1[r] > 0.0f ? acc1[r] : 0.0f;
      hT[(size_t)(r + hi * 8) * HDIM + hcol + nl]      = __float2bfloat16(v0);
      hT[(size_t)(16 + r + hi * 8) * HDIM + hcol + nl] = __float2bfloat16(v1);
    }
  }
  __syncthreads();

  // ---- layer 2: y = h @ w2 + b2, gate-scaled atomic accumulate ----
  const __hip_bfloat16* w2e = w2t + (size_t)e * DDIM * HDIM;
  for (int t = wave; t < DDIM / 16; t += 8) {
    const int dcol = t * 16;
    float bias = b2[(size_t)e * DDIM + dcol + nl];
    v8f acc0, acc1;
#pragma unroll
    for (int r = 0; r < 8; ++r) { acc0[r] = bias; acc1[r] = bias; }
    const __hip_bfloat16* bcol = w2e + (size_t)(dcol + nl) * HDIM;
    const __hip_bfloat16* arow0 = hT + (size_t)nl * HDIM;
    const __hip_bfloat16* arow1 = hT + (size_t)(16 + nl) * HDIM;
    for (int kb = 0; kb < HDIM; kb += 32) {
      v16bf Bv = *(const v16bf*)(bcol + kb + khalfB);
      __builtin_prefetch(bcol + kb + 64 + khalfB, 0, 1);
      AF A0, A1;
      A0.q[0] = *(const uint4*)(arow0 + kb + koffA);
      A0.q[1] = *(const uint4*)(arow0 + kb + 8 + koffA);
      A1.q[0] = *(const uint4*)(arow1 + kb + koffA);
      A1.q[1] = *(const uint4*)(arow1 + kb + 8 + koffA);
      acc0 = __builtin_amdgcn_wmma_f32_16x16x32_bf16(
          false, A0.v, false, Bv, (short)0, acc0, false, false);
      acc1 = __builtin_amdgcn_wmma_f32_16x16x32_bf16(
          false, A1.v, false, Bv, (short)0, acc1, false, false);
    }
#pragma unroll
    for (int r = 0; r < 8; ++r) {
      int m = r + hi * 8;
      int tok0 = toks[m];
      int tok1 = toks[16 + m];
      if (tok0 >= 0)
        atomicAdd(out + (size_t)tok0 * DDIM + dcol + nl, gts[m] * acc0[r]);
      if (tok1 >= 0)
        atomicAdd(out + (size_t)tok1 * DDIM + dcol + nl, gts[16 + m] * acc1[r]);
    }
  }
}

// ---------------- host-side launch ----------------
extern "C" void kernel_launch(void* const* d_in, const int* in_sizes, int n_in,
                              void* d_out, int out_size, void* d_ws, size_t ws_size,
                              hipStream_t stream) {
  const float* x       = (const float*)d_in[0];
  const float* noise   = (const float*)d_in[1];
  const float* w_gate  = (const float*)d_in[2];
  const float* b_gate  = (const float*)d_in[3];
  const float* w_noise = (const float*)d_in[4];
  const float* b_noise = (const float*)d_in[5];
  const float* w1      = (const float*)d_in[6];
  const float* b1      = (const float*)d_in[7];
  const float* w2      = (const float*)d_in[8];
  const float* b2      = (const float*)d_in[9];
  float* out = (float*)d_out;

  char* ws = (char*)d_ws;
  __hip_bfloat16* xb  = (__hip_bfloat16*)(ws + OFF_XB);
  __hip_bfloat16* w1t = (__hip_bfloat16*)(ws + OFF_W1T);
  __hip_bfloat16* w2t = (__hip_bfloat16*)(ws + OFF_W2T);
  int*   cnt  = (int*)(ws + OFF_CNT);
  int*   list = (int*)(ws + OFF_LIST);
  float* gv   = (float*)(ws + OFF_GVAL);

  moe_zero_kernel<<<512, 256, 0, stream>>>(out, cnt);

  moe_transpose_bf16_kernel<<<dim3(HDIM / 32, DDIM / 32, NEXP), dim3(32, 8), 0, stream>>>(
      w1, w1t, DDIM, HDIM);
  moe_transpose_bf16_kernel<<<dim3(DDIM / 32, HDIM / 32, NEXP), dim3(32, 8), 0, stream>>>(
      w2, w2t, HDIM, DDIM);

  moe_router_kernel<<<NTOK / 8, 256, 0, stream>>>(
      x, noise, w_gate, b_gate, w_noise, b_noise, xb, cnt, list, gv);

  size_t smem = (size_t)(MTILE * HDIM + MTILE * DDIM) * 2 + MTILE * 4 + MTILE * 4;
  moe_expert_kernel<<<dim3(NTOK / MTILE, NEXP), 256, smem, stream>>>(
      xb, w1t, w2t, b1, b2, cnt, list, gv, out);
}
